// GraphAttentionLayer_30726196035720
// MI455X (gfx1250) — matrix operand order repaired and checked by reading, compiled, not run on hardware
//
#include <hip/hip_runtime.h>

typedef __bf16 bf16_t;
typedef bf16_t v16bf __attribute__((ext_vector_type(16)));
typedef float  v8f   __attribute__((ext_vector_type(8)));

#define GAT_ALPHA 0.2f
#define NEG_BIG  -9.0e15f
#define LOG2E_F   1.4426950408889634f
#define NEG_BIG_L (NEG_BIG * LOG2E_F)     // NEG_BIG in log2 domain
#define C_POS     (0.6f * LOG2E_F)        // leaky fold: L*(0.6x + 0.4|x|)
#define C_ABS     (0.4f * LOG2E_F)

#define N_NODES 8192
#define F_IN    256
#define F_OUT   64
#define JWORDS  (N_NODES / 32)   // mask words per row

static __device__ __forceinline__ bf16_t f2bf(float f) { return (bf16_t)f; }
static __device__ __forceinline__ unsigned short bf_bits(float f) {
    return __builtin_bit_cast(unsigned short, f2bf(f));
}

struct U4Pair { uint4 a, b; };
static __device__ __forceinline__ v16bf load_v16bf(const void* p) {
    const uint4* q = (const uint4*)p;
    U4Pair t{q[0], q[1]};
    return __builtin_bit_cast(v16bf, t);
}

// log2-domain masked leaky-relu logit: L * leaky_relu(x)  (monotone in x)
static __device__ __forceinline__ float logit_l2(float x) {
    return fmaf(C_POS, x, C_ABS * fabsf(x));
}

// ---------------------------------------------------------------------------
// Kernel 1: Wh = h @ W (WMMA bf16), Wh1 = Wh@a1, Wh2 = Wh@a2,
//           WhT = bf16 transpose of Wh ([F_OUT][N]) for phase-3 B fragments.
// ---------------------------------------------------------------------------
__global__ __launch_bounds__(256)
void gat_gemm1(const float* __restrict__ h, const float* __restrict__ W,
               const float* __restrict__ av,
               float* __restrict__ Wh1, float* __restrict__ Wh2,
               unsigned short* __restrict__ WhT)
{
    // group g = (kt*4+f)*32 + hh*16 + c : 16 contiguous bf16, element i holds
    // W[(kt*32 + hh*8 + (i < 8 ? i : i + 8))][f*16 + c]  (one lane's B-fragment half)
    __shared__ alignas(16) unsigned short Wswz[F_IN * F_OUT];  // 32 KB
    const int tid = threadIdx.x;

    #pragma unroll
    for (int q = 0; q < 4; ++q) {
        int g  = tid * 4 + q;
        int c  = g & 15;
        int hh = (g >> 4) & 1;
        int f  = (g >> 5) & 3;
        int kt = g >> 7;
        const float* wp = W + (size_t)(kt * 32 + hh * 8) * F_OUT + f * 16 + c;
        union { unsigned short u16[16]; uint4 v[2]; } pk;
        #pragma unroll
        for (int i = 0; i < 16; ++i) pk.u16[i] = bf_bits(wp[(i < 8 ? i : i + 8) * F_OUT]);
        uint4* dst = (uint4*)(Wswz + g * 16);
        dst[0] = pk.v[0];
        dst[1] = pk.v[1];
    }
    __syncthreads();

    const int wave = tid >> 5, lane = tid & 31;
    const int tile = blockIdx.x * 8 + wave;          // 0..511
    const int r0   = tile * 16;
    const int hi   = lane >> 4;
    const int m15  = lane & 15;
    const int kb   = hi * 8;
    const int kb16 = hi * 16;

    v8f acc[4];
    #pragma unroll
    for (int f = 0; f < 4; ++f)
        #pragma unroll
        for (int i = 0; i < 8; ++i) acc[f][i] = 0.0f;

    const float* hrow = h + (size_t)(r0 + m15) * F_IN;

    #pragma unroll
    for (int kt = 0; kt < F_IN / 32; ++kt) {
        const float* hp = hrow + kt * 32;
        float4 f0 = *(const float4*)(hp + kb);
        float4 f1 = *(const float4*)(hp + kb + 4);
        float4 f2 = *(const float4*)(hp + 16 + kb);
        float4 f3 = *(const float4*)(hp + 16 + kb + 4);
        v16bf afrag;
        afrag[0]  = f2bf(f0.x); afrag[1]  = f2bf(f0.y);
        afrag[2]  = f2bf(f0.z); afrag[3]  = f2bf(f0.w);
        afrag[4]  = f2bf(f1.x); afrag[5]  = f2bf(f1.y);
        afrag[6]  = f2bf(f1.z); afrag[7]  = f2bf(f1.w);
        afrag[8]  = f2bf(f2.x); afrag[9]  = f2bf(f2.y);
        afrag[10] = f2bf(f2.z); afrag[11] = f2bf(f2.w);
        afrag[12] = f2bf(f3.x); afrag[13] = f2bf(f3.y);
        afrag[14] = f2bf(f3.z); afrag[15] = f2bf(f3.w);

        #pragma unroll
        for (int f = 0; f < 4; ++f) {
            v16bf bfrag =
                load_v16bf(Wswz + ((kt * 4 + f) * 32 + kb16 + m15) * 16);
            acc[f] = __builtin_amdgcn_wmma_f32_16x16x32_bf16(
                false, afrag, false, bfrag, (short)0, acc[f], false, false);
        }
    }

    float s1[8], s2[8];
    #pragma unroll
    for (int i = 0; i < 8; ++i) { s1[i] = 0.0f; s2[i] = 0.0f; }

    #pragma unroll
    for (int f = 0; f < 4; ++f) {
        float a1 = av[f * 16 + m15];
        float a2 = av[F_OUT + f * 16 + m15];
        union { unsigned short u16[8]; uint4 v; } pk;
        #pragma unroll
        for (int i = 0; i < 8; ++i) {
            float c = acc[f][i];
            s1[i] += c * a1;
            s2[i] += c * a2;
            pk.u16[i] = bf_bits(c);
        }
        *(uint4*)(WhT + (size_t)(f * 16 + m15) * N_NODES + r0 + 8 * hi) = pk.v;
    }

    #pragma unroll
    for (int i = 0; i < 8; ++i) {
        #pragma unroll
        for (int msk = 1; msk <= 8; msk <<= 1) {
            s1[i] += __shfl_xor(s1[i], msk, 32);
            s2[i] += __shfl_xor(s2[i], msk, 32);
        }
    }
    if (m15 == 0) {
        #pragma unroll
        for (int i = 0; i < 8; ++i) {
            Wh1[r0 + 8 * hi + i] = s1[i];
            Wh2[r0 + 8 * hi + i] = s2[i];
        }
    }
}

// ---------------------------------------------------------------------------
// Kernel 2: per-row softmax stats in log2 domain (+ optional 1-bit mask).
// Stores Bias[i] = -max_row - log2(sum_row). One exp per adj element.
// ---------------------------------------------------------------------------
template <bool EMIT_MASK>
__global__ __launch_bounds__(256)
void gat_rowstats(const int* __restrict__ adj, const float* __restrict__ Wh1,
                  const float* __restrict__ Wh2,
                  float* __restrict__ Bias, unsigned* __restrict__ Mask)
{
    __shared__ float red[256];
    const int i = blockIdx.x;
    const int tid = threadIdx.x;
    const float w1 = Wh1[i];
    const int* arow = adj + (size_t)i * N_NODES + tid * 32;
    const float* w2p = Wh2 + tid * 32;

    float ev[32];
    unsigned mword = 0;
    #pragma unroll
    for (int s = 0; s < 8; ++s) {
        int4   a4 = *(const int4*)(arow + s * 4);
        float4 w4 = *(const float4*)(w2p + s * 4);
        float t;
        t = logit_l2(w1 + w4.x); ev[s*4+0] = (a4.x > 0) ? t : NEG_BIG_L;
        t = logit_l2(w1 + w4.y); ev[s*4+1] = (a4.y > 0) ? t : NEG_BIG_L;
        t = logit_l2(w1 + w4.z); ev[s*4+2] = (a4.z > 0) ? t : NEG_BIG_L;
        t = logit_l2(w1 + w4.w); ev[s*4+3] = (a4.w > 0) ? t : NEG_BIG_L;
        if (EMIT_MASK) {
            mword |= (a4.x > 0 ? 1u : 0u) << (s * 4 + 0);
            mword |= (a4.y > 0 ? 1u : 0u) << (s * 4 + 1);
            mword |= (a4.z > 0 ? 1u : 0u) << (s * 4 + 2);
            mword |= (a4.w > 0 ? 1u : 0u) << (s * 4 + 3);
        }
    }
    if (EMIT_MASK) Mask[(size_t)i * JWORDS + tid] = mword;

    float m = -3.0e38f;
    #pragma unroll
    for (int k = 0; k < 32; ++k) m = fmaxf(m, ev[k]);

    red[tid] = m;
    __syncthreads();
    for (int st = 128; st > 0; st >>= 1) {
        if (tid < st) red[tid] = fmaxf(red[tid], red[tid + st]);
        __syncthreads();
    }
    const float mrow = red[0];
    __syncthreads();

    float l = 0.0f;
    #pragma unroll
    for (int k = 0; k < 32; ++k) l += __builtin_amdgcn_exp2f(ev[k] - mrow);

    red[tid] = l;
    __syncthreads();
    for (int st = 128; st > 0; st >>= 1) {
        if (tid < st) red[tid] += red[tid + st];
        __syncthreads();
    }
    if (tid == 0) Bias[i] = -mrow - log2f(red[0]);
}

// ---------------------------------------------------------------------------
// Kernel 3: fused masked-softmax attention @ Wh via WMMA bf16 + ELU.
// 32 output rows (2 WMMA i-tiles) per block so each B fragment feeds 2 WMMAs.
// 8 waves split the j-range; partials reduced through LDS.
// p = exp2( fma(0.6L, x, fma(0.4L, |x|, bias)) ), mask -> hoisted argMask.
// ---------------------------------------------------------------------------
template <bool USE_MASK>
__global__ __launch_bounds__(256)
void gat_attn(const int* __restrict__ adj, const unsigned* __restrict__ Mask,
              const unsigned short* __restrict__ WhT,
              const float* __restrict__ Wh1, const float* __restrict__ Wh2,
              const float* __restrict__ Bias, float* __restrict__ out)
{
    __shared__ float wacc[8][32 * F_OUT];  // 64 KB per-wave partials

    const int tid  = threadIdx.x;
    const int wave = tid >> 5, lane = tid & 31;
    const int r0   = blockIdx.x * 32;

    const int hi   = lane >> 4;
    const int m15  = lane & 15;
    const int kb   = hi * 8;
    const int kb16 = hi * 16;
    const int ir0  = r0 + m15;           // A row, i-tile 0
    const int ir1  = r0 + 16 + m15;      // A row, i-tile 1

    const float w10 = Wh1[ir0],  w11 = Wh1[ir1];
    const float b0  = Bias[ir0], b1  = Bias[ir1];
    const float am0 = NEG_BIG_L + b0;    // masked exp2 argument
    const float am1 = NEG_BIG_L + b1;
    const unsigned* mr0 = Mask + (size_t)ir0 * JWORDS;
    const unsigned* mr1 = Mask + (size_t)ir1 * JWORDS;
    const int* ar0 = adj + (size_t)ir0 * N_NODES;
    const int* ar1 = adj + (size_t)ir1 * N_NODES;

    v8f acc0[4], acc1[4];
    #pragma unroll
    for (int f = 0; f < 4; ++f)
        #pragma unroll
        for (int i = 0; i < 8; ++i) { acc0[f][i] = 0.0f; acc1[f][i] = 0.0f; }

    for (int jt = wave; jt < JWORDS; jt += 8) {
        const int j0 = jt * 32;

        float w2v[16];
        {
            float4 wa = *(const float4*)(Wh2 + j0 + kb);
            float4 wb = *(const float4*)(Wh2 + j0 + kb + 4);
            float4 wc = *(const float4*)(Wh2 + j0 + 16 + kb);
            float4 wd = *(const float4*)(Wh2 + j0 + 16 + kb + 4);
            w2v[0]=wa.x;  w2v[1]=wa.y;  w2v[2]=wa.z;  w2v[3]=wa.w;
            w2v[4]=wb.x;  w2v[5]=wb.y;  w2v[6]=wb.z;  w2v[7]=wb.w;
            w2v[8]=wc.x;  w2v[9]=wc.y;  w2v[10]=wc.z; w2v[11]=wc.w;
            w2v[12]=wd.x; w2v[13]=wd.y; w2v[14]=wd.z; w2v[15]=wd.w;
        }

        unsigned mw0 = 0, mw1 = 0;
        int af0[16], af1[16];
        if (USE_MASK) {
            __builtin_prefetch(mr0 + jt + 8, 0, 0);
            __builtin_prefetch(mr1 + jt + 8, 0, 0);
            mw0 = mr0[jt];
            mw1 = mr1[jt];
        } else {
            int4 q;
            q = *(const int4*)(ar0 + j0 + kb);          af0[0]=q.x;  af0[1]=q.y;  af0[2]=q.z;  af0[3]=q.w;
            q = *(const int4*)(ar0 + j0 + kb + 4);      af0[4]=q.x;  af0[5]=q.y;  af0[6]=q.z;  af0[7]=q.w;
            q = *(const int4*)(ar0 + j0 + 16 + kb);     af0[8]=q.x;  af0[9]=q.y;  af0[10]=q.z; af0[11]=q.w;
            q = *(const int4*)(ar0 + j0 + 16 + kb + 4); af0[12]=q.x; af0[13]=q.y; af0[14]=q.z; af0[15]=q.w;
            q = *(const int4*)(ar1 + j0 + kb);          af1[0]=q.x;  af1[1]=q.y;  af1[2]=q.z;  af1[3]=q.w;
            q = *(const int4*)(ar1 + j0 + kb + 4);      af1[4]=q.x;  af1[5]=q.y;  af1[6]=q.z;  af1[7]=q.w;
            q = *(const int4*)(ar1 + j0 + 16 + kb);     af1[8]=q.x;  af1[9]=q.y;  af1[10]=q.z; af1[11]=q.w;
            q = *(const int4*)(ar1 + j0 + 16 + kb + 4); af1[12]=q.x; af1[13]=q.y; af1[14]=q.z; af1[15]=q.w;
        }

        v16bf p0, p1;
        #pragma unroll
        for (int i = 0; i < 16; ++i) {
            const int sh = (i < 8) ? (kb + i) : (8 + kb + i);  // 16+kb+(i-8)
            bool on0, on1;
            if (USE_MASK) {
                on0 = ((mw0 >> sh) & 1u) != 0u;
                on1 = ((mw1 >> sh) & 1u) != 0u;
            } else {
                on0 = af0[i] > 0;
                on1 = af1[i] > 0;
            }
            float x0 = w10 + w2v[i];
            float x1 = w11 + w2v[i];
            float g0 = fmaf(C_POS, x0, fmaf(C_ABS, fabsf(x0), b0));
            float g1 = fmaf(C_POS, x1, fmaf(C_ABS, fabsf(x1), b1));
            p0[i] = f2bf(__builtin_amdgcn_exp2f(on0 ? g0 : am0));
            p1[i] = f2bf(__builtin_amdgcn_exp2f(on1 ? g1 : am1));
        }

        #pragma unroll
        for (int f = 0; f < 4; ++f) {
            const unsigned short* wtp = WhT + (size_t)(f * 16 + m15) * N_NODES + j0;
            U4Pair bp{*(const uint4*)(wtp + kb), *(const uint4*)(wtp + 16 + kb)};
            v16bf bfrag = __builtin_bit_cast(v16bf, bp);
            acc0[f] = __builtin_amdgcn_wmma_f32_16x16x32_bf16(
                false, p0, false, bfrag, (short)0, acc0[f], false, false);
            acc1[f] = __builtin_amdgcn_wmma_f32_16x16x32_bf16(
                false, p1, false, bfrag, (short)0, acc1[f], false, false);
        }
    }

    #pragma unroll
    for (int f = 0; f < 4; ++f)
        #pragma unroll
        for (int i = 0; i < 8; ++i) {
            wacc[wave][(i + 8 * hi) * F_OUT + f * 16 + m15]      = acc0[f][i];
            wacc[wave][(16 + i + 8 * hi) * F_OUT + f * 16 + m15] = acc1[f][i];
        }
    __syncthreads();

    // Reduce 8 wave partials over 2048 outputs (8 per thread), ELU, store.
    #pragma unroll
    for (int q = 0; q < 2; ++q) {
        const int base = tid * 8 + q * 4;
        float4 s = make_float4(0.f, 0.f, 0.f, 0.f);
        #pragma unroll
        for (int w = 0; w < 8; ++w) {
            float4 v = *(const float4*)(&wacc[w][base]);
            s.x += v.x; s.y += v.y; s.z += v.z; s.w += v.w;
        }
        s.x = (s.x > 0.f) ? s.x : expm1f(s.x);
        s.y = (s.y > 0.f) ? s.y : expm1f(s.y);
        s.z = (s.z > 0.f) ? s.z : expm1f(s.z);
        s.w = (s.w > 0.f) ? s.w : expm1f(s.w);
        *(float4*)(out + (size_t)r0 * F_OUT + base) = s;
    }
}

// ---------------------------------------------------------------------------
extern "C" void kernel_launch(void* const* d_in, const int* in_sizes, int n_in,
                              void* d_out, int out_size, void* d_ws, size_t ws_size,
                              hipStream_t stream) {
    const float* h   = (const float*)d_in[0];
    const int*   adj = (const int*)d_in[1];
    const float* W   = (const float*)d_in[2];
    const float* av  = (const float*)d_in[3];
    float* out = (float*)d_out;

    char* ws = (char*)d_ws;
    float*          Wh1  = (float*)(ws);                 // 32 KB
    float*          Wh2  = (float*)(ws + (32u << 10));   // 32 KB
    float*          Bias = (float*)(ws + (64u << 10));   // 32 KB
    unsigned short* WhT  = (unsigned short*)(ws + (128u << 10)); // 1 MB bf16
    const size_t maskOff = (128u << 10) + (size_t)F_OUT * N_NODES * 2;
    unsigned*       Mask = (unsigned*)(ws + maskOff);    // 8 MB bitmask
    const bool use_mask =
        ws_size >= maskOff + (size_t)N_NODES * JWORDS * sizeof(unsigned);

    gat_gemm1<<<N_NODES / 16 / 8, 256, 0, stream>>>(h, W, av, Wh1, Wh2, WhT);
    if (use_mask) {
        gat_rowstats<true><<<N_NODES, 256, 0, stream>>>(adj, Wh1, Wh2,
                                                        Bias, Mask);
        gat_attn<true><<<N_NODES / 32, 256, 0, stream>>>(adj, Mask, WhT, Wh1,
                                                         Wh2, Bias, out);
    } else {
        gat_rowstats<false><<<N_NODES, 256, 0, stream>>>(adj, Wh1, Wh2,
                                                         Bias, nullptr);
        gat_attn<false><<<N_NODES / 32, 256, 0, stream>>>(adj, nullptr, WhT,
                                                          Wh1, Wh2, Bias, out);
    }
}
